// Decoder_76390288326822
// MI455X (gfx1250) — compile-verified, matrix operand order checked by reading, not run on hardware
//
#include <hip/hip_runtime.h>
#include <hip/hip_bf16.h>
#include <stdint.h>

#define DDIM 512
#define NDIM 2048          // 4*D
#define KSTEPS 24
#define BTOT 8192
#define MT 32              // batch rows per workgroup (2 row tiles of 16)
#define THREADS 256

typedef __attribute__((ext_vector_type(16))) __bf16 v16bf;
typedef __attribute__((ext_vector_type(8)))  float  v8f;

__device__ __forceinline__ unsigned short f2bf_bits(float f) {
    union { float f; uint32_t u; } x; x.f = f;
    uint32_t r = (x.u + 0x7FFFu + ((x.u >> 16) & 1u)) >> 16;   // RNE
    return (unsigned short)r;
}
__device__ __forceinline__ float bf2f(unsigned short s) {
    union { uint32_t u; float f; } x; x.u = ((uint32_t)s) << 16; return x.f;
}
// Branch-free transcendentals: v_exp_f32 + v_rcp_f32, no exec-mask splits.
__device__ __forceinline__ float sigmoid_fast(float x) {
    return __builtin_amdgcn_rcpf(1.0f + __expf(-x));
}
__device__ __forceinline__ float tanh_fast(float x) {
    // 1 - 2/(1+e^{2x}); saturates to +-1 as exp -> inf/0, NaN-free.
    return 1.0f - 2.0f * __builtin_amdgcn_rcpf(1.0f + __expf(2.0f * x));
}

// ---------------------------------------------------------------------------
// Prologue: convert Wu (D x 4D, fp32 row-major) into bf16 fragments laid out
// exactly as v_wmma_f32_16x16x32_bf16 wants its B matrix per lane:
//   fragment f = jg*16 + kc; per lane 16 bf16 (=8 VGPRs), 32B contiguous.
// ---------------------------------------------------------------------------
__global__ void pack_wu_kernel(const float* __restrict__ Wu,
                               unsigned short* __restrict__ wuP) {
    int tid  = blockIdx.x * blockDim.x + threadIdx.x;   // 128*16*32 threads
    int lane = tid & 31;
    int f    = tid >> 5;
    int kc   = f & 15;
    int jg   = f >> 4;                 // 0..127 over 2048 cols
    int n    = jg * 16 + (lane & 15);
    int base = (lane < 16) ? 0 : 8;

    uint32_t w[8];
    #pragma unroll
    for (int v = 0; v < 8; ++v) {
        int k_even = (v < 4) ? (base + 2 * v) : (16 + base + 2 * (v - 4));
        int k = kc * 32 + k_even;
        uint32_t lo = f2bf_bits(Wu[(size_t)k       * NDIM + n]);
        uint32_t hi = f2bf_bits(Wu[(size_t)(k + 1) * NDIM + n]);
        w[v] = lo | (hi << 16);
    }
    uint4* dst = (uint4*)(wuP + ((size_t)(f * 32 + lane)) * 16);
    dst[0] = *(uint4*)&w[0];
    dst[1] = *(uint4*)&w[4];
}

// ---------------------------------------------------------------------------
// Persistent autoregressive LSTM decode. One workgroup owns MT=32 batch rows
// (two 16-row WMMA tiles) for all 24 steps. Each B fragment loaded from L2 is
// reused by 2 wmma (one per row tile) -> halved L2 traffic per FLOP. Cell
// state lives in VGPRs; hidden state double-buffered bf16 in LDS.
// ---------------------------------------------------------------------------
__global__ __launch_bounds__(THREADS, 1)
void lstm_decode_kernel(const float* __restrict__ x0,
                        const float* __restrict__ h0,
                        const float* __restrict__ c0,
                        const float* __restrict__ Wx,
                        const float* __restrict__ bvec,
                        const float* __restrict__ w1, const float* __restrict__ b1,
                        const float* __restrict__ w2, const float* __restrict__ b2,
                        const unsigned short* __restrict__ wuP,
                        float* __restrict__ out) {
    __shared__ unsigned short hbuf[2][MT * DDIM];   // 2 x 32 KB, bf16 row-major
    __shared__ float xbuf[MT];
    __shared__ float WxS[NDIM];                     // 8 KB
    __shared__ float bS[NDIM];                      // 8 KB
    __shared__ float w1S[DDIM];                     // 2 KB
    __shared__ float w2S[DDIM];                     // 2 KB

    const int tid    = threadIdx.x;
    const int wave   = tid >> 5;
    const int lane   = tid & 31;
    const int lanelo = lane & 15;
    const int hi     = lane >> 4;        // 0 or 1 (lane half)
    const int row0   = blockIdx.x * MT;

    for (int i = tid; i < NDIM; i += THREADS) { WxS[i] = Wx[i]; bS[i] = bvec[i]; }
    for (int i = tid; i < DDIM; i += THREADS) { w1S[i] = w1[i]; w2S[i] = w2[i]; }
    for (int i = tid; i < MT * DDIM; i += THREADS)
        hbuf[0][i] = f2bf_bits(h0[(size_t)(row0 + (i >> 9)) * DDIM + (i & 511)]);
    if (tid < MT) xbuf[tid] = x0[row0 + tid];

    // Cell state: 4 col tiles (j = wave*4 + bi) x 2 row tiles, C layout:
    // element v -> row M = r*16 + v + 8*hi, col N = j*16 + lanelo
    float creg[4][2][8];
    #pragma unroll
    for (int bi = 0; bi < 4; ++bi) {
        const int j = wave * 4 + bi;
        #pragma unroll
        for (int r = 0; r < 2; ++r)
            #pragma unroll
            for (int v = 0; v < 8; ++v)
                creg[bi][r][v] =
                    c0[(size_t)(row0 + r * 16 + v + 8 * hi) * DDIM + j * 16 + lanelo];
    }
    const float b1v = b1[0], b2v = b2[0];
    __syncthreads();

    #pragma unroll 1
    for (int k = 0; k < KSTEPS; ++k) {
        const int cur = k & 1, nxt = cur ^ 1;

        // x feedback values (per row tile), hoisted out of the tile loop.
        float xv[2][8];
        #pragma unroll
        for (int r = 0; r < 2; ++r)
            #pragma unroll
            for (int v = 0; v < 8; ++v)
                xv[r][v] = xbuf[r * 16 + v + 8 * hi];

        #pragma unroll 1
        for (int bi = 0; bi < 4; ++bi) {
            const int j   = wave * 4 + bi;
            const int col = j * 16 + lanelo;

            // Per-lane base pointers: all hot-loop addresses are compile-time
            // immediates off these (A: r*16384 + kc*64 bytes in DS offset;
            // B: g*512KB + kc*1KB in the 24-bit global IOFFSET).
            const uint32_t* abase =
                (const uint32_t*)&hbuf[cur][0] + lanelo * (DDIM / 2) + 4 * hi;
            const unsigned short* wbase =
                wuP + (size_t)j * 8192 + (size_t)lane * 16;

            // Accumulator init: fold in x (rank-1) and bias, per C-layout elem
            v8f acc[2][4];
            #pragma unroll
            for (int g = 0; g < 4; ++g) {
                const float wxc = WxS[g * DDIM + col];
                const float bc  = bS[g * DDIM + col];
                #pragma unroll
                for (int r = 0; r < 2; ++r)
                    #pragma unroll
                    for (int v = 0; v < 8; ++v)
                        acc[r][g][v] = xv[r][v] * wxc + bc;
            }

            // K reduction: 16 chunks of 32. Per chunk: 2 A frags (row tiles,
            // ds_load_b128 x4) + 4 B frags (global_load_b128 x8) -> 8 wmma.
            // Each B fragment feeds both row tiles.
            #pragma unroll
            for (int kc = 0; kc < 16; ++kc) {
                union AV { v16bf v; uint4 q[2]; };
                AV a[2];
                #pragma unroll
                for (int r = 0; r < 2; ++r) {
                    a[r].q[0] = *(const uint4*)(abase + r * 4096 + kc * 16);
                    a[r].q[1] = *(const uint4*)(abase + r * 4096 + kc * 16 + 8);
                }
                #pragma unroll
                for (int g = 0; g < 4; ++g) {
                    const v16bf bf =
                        *(const v16bf*)(wbase + g * 262144 + kc * 512);
                    #pragma unroll
                    for (int r = 0; r < 2; ++r)
                        acc[r][g] = __builtin_amdgcn_wmma_f32_16x16x32_bf16(
                            false, a[r].v, false, bf,
                            (short)0, acc[r][g], false, false);
                }
            }

            // Gates + cell update in registers; new h -> other LDS buffer
            #pragma unroll
            for (int r = 0; r < 2; ++r)
                #pragma unroll
                for (int v = 0; v < 8; ++v) {
                    float ig = sigmoid_fast(acc[r][0][v]);
                    float fg = sigmoid_fast(acc[r][1][v]);
                    float gg = tanh_fast(acc[r][2][v]);
                    float og = sigmoid_fast(acc[r][3][v]);
                    float cn = fg * creg[bi][r][v] + ig * gg;
                    creg[bi][r][v] = cn;
                    float hn = og * tanh_fast(cn);
                    hbuf[nxt][(r * 16 + v + 8 * hi) * DDIM + col] = f2bf_bits(hn);
                }
        }
        __syncthreads();

        // y-phase: 32 rows x 8 lanes; each lane sums 64 cols, xor-reduce
        {
            const int row = tid >> 3;
            const int seg = tid & 7;
            float s1 = 0.f, s2 = 0.f;
            #pragma unroll 8
            for (int c = seg * 64; c < seg * 64 + 64; ++c) {
                float hv = bf2f(hbuf[nxt][row * DDIM + c]);
                s1 += hv * w1S[c];
                s2 += hv * w2S[c];
            }
            #pragma unroll
            for (int off = 1; off < 8; off <<= 1) {
                s1 += __shfl_xor(s1, off, 32);
                s2 += __shfl_xor(s2, off, 32);
            }
            if (seg == 0) {
                float y1 = sigmoid_fast(s1 + b1v);
                float t  = s2 + b2v;
                float y2 = (t > 0.f) ? t : (__expf(t) - 1.0f);
                const int gb = row0 + row;
                out[(size_t)gb * KSTEPS + k] = y1;
                out[(size_t)(BTOT * KSTEPS) + (size_t)gb * KSTEPS + k] = y2;
                xbuf[row] = y1;   // autoregressive feedback
            }
        }
        __syncthreads();
    }
}

extern "C" void kernel_launch(void* const* d_in, const int* in_sizes, int n_in,
                              void* d_out, int out_size, void* d_ws, size_t ws_size,
                              hipStream_t stream) {
    const float* x0  = (const float*)d_in[0];   // initial (B,1,1)
    const float* h0  = (const float*)d_in[1];   // encoder_hidden (B,D)
    const float* c0  = (const float*)d_in[2];   // encoder_cell (B,D)
    const float* Wx  = (const float*)d_in[3];   // (1,4D)
    const float* Wu  = (const float*)d_in[4];   // (D,4D)
    const float* b   = (const float*)d_in[5];   // (4D)
    const float* w1  = (const float*)d_in[6];
    const float* b1  = (const float*)d_in[7];
    const float* w2  = (const float*)d_in[8];
    const float* b2  = (const float*)d_in[9];
    unsigned short* wuP = (unsigned short*)d_ws;   // 2 MB bf16 fragments

    pack_wu_kernel<<<(128 * 16 * 32) / 256, 256, 0, stream>>>(Wu, wuP);
    lstm_decode_kernel<<<BTOT / MT, THREADS, 0, stream>>>(
        x0, h0, c0, Wx, b, w1, b1, w2, b2, wuP, (float*)d_out);
}
